// NeuralCDE_38422777430610
// MI455X (gfx1250) — matrix operand order repaired and checked by reading, not compiled, not run on hardware
//
#include <hip/hip_runtime.h>
#include <hip/hip_bf16.h>
#include <math.h>

// Problem constants (from reference): B=1024, T=256, D=8, H=128
#define BB 1024
#define TT 256
#define DD 8
#define HH 128

typedef __attribute__((ext_vector_type(16))) __bf16 v16bf;
typedef __attribute__((ext_vector_type(8)))  float  v8f;
typedef __attribute__((ext_vector_type(4)))  unsigned int u32x4;
typedef __attribute__((ext_vector_type(8)))  int i32x8;
typedef __attribute__((ext_vector_type(4)))  int i32x4;

// Load a 16x32 bf16 WMMA fragment as two 16-byte chunks (per CDNA5 VGPR layout).
__device__ __forceinline__ v16bf load_frag2(const __bf16* p0, const __bf16* p1) {
    union { v16bf v; u32x4 q[2]; } u;
    u.q[0] = *(const u32x4*)p0;
    u.q[1] = *(const u32x4*)p1;
    return u.v;
}

__device__ __forceinline__ v8f wmma_bf16(v16bf a, v16bf b, v8f c) {
    return __builtin_amdgcn_wmma_f32_16x16x32_bf16(
        /*neg_a=*/false, a, /*neg_b=*/false, b,
        /*c_mod=*/(short)0, c, /*reuse_a=*/false, /*reuse_b=*/false);
}

// ---------------------------------------------------------------------------
// Prep kernel: build bf16 transposed weights in workspace.
// ---------------------------------------------------------------------------
__global__ void cde_prep(const float* __restrict__ W0, const float* __restrict__ W1,
                         const float* __restrict__ W2,
                         __bf16* __restrict__ w0t, __bf16* __restrict__ w1t,
                         __bf16* __restrict__ w2t, float* __restrict__ wtrow) {
    int i = blockIdx.x * 256 + threadIdx.x;
    if (i < 16384) {
        int n = i >> 7, k = i & 127;
        w0t[i] = (__bf16)W0[k * 128 + n];
    } else if (i < 32768) {
        int j = i - 16384; int n = j >> 7, k = j & 127;
        w1t[j] = (__bf16)W1[k * 128 + n];
    } else if (i < 163840) {
        int j = i - 32768; int n = j >> 7, k = j & 127;
        w2t[j] = (__bf16)W2[k * 1024 + n];
    } else if (i < 163968) {
        int n = i - 163840;
        wtrow[n] = W0[128 * 128 + n];
    }
}

// ---------------------------------------------------------------------------
// Persistent rollout kernel: one block per 16-row batch tile, 8 wave32s.
// W2 (hot GEMM3 operand, 32 of 40 WMMAs/step) lives in LDS for the rollout,
// staged once via the Tensor Data Mover; W0/W1 fragments end up hoisted into
// persistent VGPRs by the compiler (loop-invariant L2 loads).
// ---------------------------------------------------------------------------
#define LDK 136   // padded K stride (bf16 elems): 272B = 17*16B -> aligned, bank-skewed

__global__ __launch_bounds__(256) void cde_main(
    const float* __restrict__ x, const float* __restrict__ W_in,
    const float* __restrict__ b_in, const float* __restrict__ b0v,
    const float* __restrict__ b1v, const float* __restrict__ b2v,
    const __bf16* __restrict__ w0t, const __bf16* __restrict__ w1t,
    const __bf16* __restrict__ w2t, const float* __restrict__ wtrow,
    float* __restrict__ out) {

    __shared__ __align__(16) __bf16 sW2[1024 * LDK]; // W2 transposed [n][k] (272 KB)
    __shared__ __align__(16) float  sZ[16 * 128];    // recurrent state, fp32
    __shared__ __align__(16) __bf16 sZb[16 * LDK];   // z as bf16 (A operand)
    __shared__ __align__(16) __bf16 sH1[16 * LDK];
    __shared__ __align__(16) __bf16 sH2[16 * LDK];
    __shared__ __align__(16) float  sDx[16 * 8];
    __shared__ float sB0[128], sB1[128], sWt[128], sB2[1024];

    const int tid  = threadIdx.x;
    const int wave = tid >> 5;
    const int lane = tid & 31;
    const int brow = blockIdx.x * 16;          // batch tile base

    // ---- one-time: stage all of W2 into LDS ----
#if __has_builtin(__builtin_amdgcn_tensor_load_to_lds)
    if (wave == 0) {
        // Tensor DMA descriptor (D#): 2D tensor 1024(n) x 128(k) bf16, full tile,
        // LDS padding 4 DWORDs every 64 DWORDs -> row stride 272B (= LDK*2).
        const unsigned lds_base = (unsigned)(size_t)(&sW2[0]);     // LDS offset
        const unsigned long long ga = (unsigned long long)(size_t)w2t;
        u32x4 g0;
        g0[0] = 1u;                                  // count=1 (valid user D#)
        g0[1] = lds_base;                            // lds_addr
        g0[2] = (unsigned)(ga & 0xFFFFFFFFu);        // global_addr[31:0]
        g0[3] = (unsigned)((ga >> 32) & 0x01FFFFFFu) // global_addr[56:32]
              | (2u << 30);                          // type = 2 ("image")
        i32x8 g1;
        g1[0] = (int)((1u << 16)      // data_size = 1 -> 2 bytes
                    | (1u << 20)      // pad_enable
                    | (5u << 22)      // pad_interval: 64 DWORDs (256B row)
                    | (3u << 25));    // pad_amount: 4 DWORDs (16B) -> stride 272B
        g1[1] = (int)(128u << 16);    // tensor_dim0 = 128 (k)
        g1[2] = (int)(1024u << 16);   // tensor_dim1 = 1024 (n)
        g1[3] = (int)(128u << 16);    // tile_dim0 = 128
        g1[4] = 1024;                 // tile_dim1 = 1024
        g1[5] = 128;                  // tensor_dim0_stride = 128 elements
        g1[6] = 0;
        g1[7] = 0;
        i32x4 gz = {0, 0, 0, 0};
#if __clang_major__ >= 23
        i32x8 gz8 = {0, 0, 0, 0, 0, 0, 0, 0};
        __builtin_amdgcn_tensor_load_to_lds(g0, g1, gz, gz, gz8, 0);
#else
        __builtin_amdgcn_tensor_load_to_lds(g0, g1, gz, gz, 0);
#endif
        __builtin_amdgcn_s_wait_tensorcnt(0);
    }
#else
    for (int i = tid; i < 1024 * 16; i += 256) {        // 8 bf16 per chunk
        int n = i >> 4;
        int c = (i & 15) * 8;
        *(u32x4*)&sW2[n * LDK + c] = *(const u32x4*)&w2t[n * 128 + c];
    }
#endif
    if (tid < 128) { sB0[tid] = b0v[tid]; sB1[tid] = b1v[tid]; sWt[tid] = wtrow[tid]; }
    for (int i = tid; i < 1024; i += 256) sB2[i] = b2v[i];

    // ---- z0 = x[:,0,:] @ W_in + b_in ; write out[:,0,:] ; z -> bf16 ----
    for (int i = tid; i < 16 * 128; i += 256) {
        int m = i >> 7, h = i & 127;
        const float* xr = x + (size_t)(brow + m) * TT * DD;   // t = 0
        float acc = b_in[h];
        #pragma unroll
        for (int d = 0; d < DD; ++d) acc += xr[d] * W_in[d * HH + h];
        sZ[i] = acc;
        sZb[m * LDK + h] = (__bf16)acc;
        __builtin_nontemporal_store(acc, &out[((size_t)(brow + m) * TT) * HH + h]);
    }
    if (tid < 128) {                                     // dx for step 0
        int m = tid >> 3, d = tid & 7;
        const float* xr = x + (size_t)(brow + m) * TT * DD + d;
        sDx[tid] = xr[DD] - xr[0];
    }
    __syncthreads();

    const int arow = lane & 15;                 // A: row within tile
    const int asel = (lane >> 4) * 8;           // A: K sub-chunk select
    const int bcol = lane & 15;                 // B/C: column within tile
    const int bsel = (lane >> 4) * 8;           // B: K sub-chunk select
    const int mhalf = (lane >> 4) * 8;          // C: row-half base

    for (int k = 0; k < TT - 1; ++k) {
        const float tk = (float)k * (1.0f / (float)(TT - 1));

        // ---- GEMM1: h1 = relu(z@W0' + b0 + tk*wt)  (8 n-tiles, 1 per wave) ----
        {
            const int nt = wave;
            const __bf16* w0p = w0t + (size_t)(nt * 16 + bcol) * 128;  // -> VGPR-hoisted
            v8f acc = {};
            #pragma unroll
            for (int kk = 0; kk < 4; ++kk) {
                const int kb = kk * 32;
                v16bf a = load_frag2(&sZb[arow * LDK + kb + asel],
                                     &sZb[arow * LDK + kb + asel + 16]);
                v16bf b = load_frag2(w0p + kb + bsel, w0p + kb + bsel + 16);
                acc = wmma_bf16(a, b, acc);
            }
            const int n = nt * 16 + bcol;
            const float bias = sB0[n] + tk * sWt[n];
            #pragma unroll
            for (int v = 0; v < 8; ++v) {
                float val = acc[v] + bias;
                val = val > 0.0f ? val : 0.0f;
                sH1[(mhalf + v) * LDK + n] = (__bf16)val;
            }
        }
        __syncthreads();

        // ---- GEMM2: h2 = relu(h1@W1 + b1) ----
        {
            const int nt = wave;
            const __bf16* w1p = w1t + (size_t)(nt * 16 + bcol) * 128;  // -> VGPR-hoisted
            v8f acc = {};
            #pragma unroll
            for (int kk = 0; kk < 4; ++kk) {
                const int kb = kk * 32;
                v16bf a = load_frag2(&sH1[arow * LDK + kb + asel],
                                     &sH1[arow * LDK + kb + asel + 16]);
                v16bf b = load_frag2(w1p + kb + bsel, w1p + kb + bsel + 16);
                acc = wmma_bf16(a, b, acc);
            }
            const int n = nt * 16 + bcol;
            const float bias = sB1[n];
            #pragma unroll
            for (int v = 0; v < 8; ++v) {
                float val = acc[v] + bias;
                val = val > 0.0f ? val : 0.0f;
                sH2[(mhalf + v) * LDK + n] = (__bf16)val;
            }
        }
        __syncthreads();

        // ---- GEMM3 + einsum: F = tanh(h2@W2 + b2); z += F·dx ----
        // 64 n-tiles over the 1024 columns; 8 per wave. Column n = h*8 + d.
        {
            v16bf a3[4];
            #pragma unroll
            for (int kk = 0; kk < 4; ++kk) {
                const int kb = kk * 32;
                a3[kk] = load_frag2(&sH2[arow * LDK + kb + asel],
                                    &sH2[arow * LDK + kb + asel + 16]);
            }
            for (int i = 0; i < 8; ++i) {
                const int nt = wave * 8 + i;
                const __bf16* w2p = &sW2[(size_t)(nt * 16 + bcol) * LDK];
                v8f acc = {};
                #pragma unroll
                for (int kk = 0; kk < 4; ++kk) {
                    const int kb = kk * 32;
                    v16bf b = load_frag2(w2p + kb + bsel, w2p + kb + bsel + 16);
                    acc = wmma_bf16(a3[kk], b, acc);
                }
                const int n = nt * 16 + bcol;
                const float bias = sB2[n];
                const int d = n & 7;
                const int h = n >> 3;              // h = 2*nt or 2*nt+1 per 8-lane group (lane bit 3)
                #pragma unroll
                for (int v = 0; v < 8; ++v) {
                    const int m = mhalf + v;
                    float p = tanhf(acc[v] + bias) * sDx[m * 8 + d];
                    // reduce the 8 lanes sharing (m, h)
                    p += __shfl_xor(p, 1, 32);
                    p += __shfl_xor(p, 2, 32);
                    p += __shfl_xor(p, 4, 32);
                    if ((lane & 7) == 0) sZ[m * 128 + h] += p;
                }
            }
        }
        __syncthreads();

        // ---- fused tail: write out[:,k+1,:] (NT), z->bf16, dx for step k+1 ----
        for (int i = tid; i < 16 * 128; i += 256) {
            int m = i >> 7, h = i & 127;
            float val = sZ[i];
            __builtin_nontemporal_store(val,
                &out[((size_t)(brow + m) * TT + (k + 1)) * HH + h]);
            sZb[m * LDK + h] = (__bf16)val;
        }
        if (tid < 128 && k + 1 < TT - 1) {
            int m = tid >> 3, d = tid & 7;
            const float* xr = x + ((size_t)(brow + m) * TT + (k + 1)) * DD + d;
            sDx[tid] = xr[DD] - xr[0];
            if (k + 3 < TT) __builtin_prefetch(xr + 2 * DD, 0, 0);   // next x slice
        }
        __syncthreads();
    }
}

// ---------------------------------------------------------------------------
extern "C" void kernel_launch(void* const* d_in, const int* in_sizes, int n_in,
                              void* d_out, int out_size, void* d_ws, size_t ws_size,
                              hipStream_t stream) {
    (void)in_sizes; (void)n_in; (void)out_size; (void)ws_size;
    const float* x    = (const float*)d_in[0];
    const float* W_in = (const float*)d_in[1];
    const float* b_in = (const float*)d_in[2];
    const float* W0   = (const float*)d_in[3];
    const float* b0   = (const float*)d_in[4];
    const float* W1   = (const float*)d_in[5];
    const float* b1   = (const float*)d_in[6];
    const float* W2   = (const float*)d_in[7];
    const float* b2   = (const float*)d_in[8];
    float* out = (float*)d_out;

    char* ws = (char*)d_ws;
    __bf16* w0t  = (__bf16*)(ws);            // 32768 B
    __bf16* w1t  = (__bf16*)(ws + 32768);    // 32768 B
    __bf16* w2t  = (__bf16*)(ws + 65536);    // 262144 B
    float*  wtrw = (float*)(ws + 327680);    // 512 B

    cde_prep<<<(163968 + 255) / 256, 256, 0, stream>>>(W0, W1, W2, w0t, w1t, w2t, wtrw);
    cde_main<<<BB / 16, 256, 0, stream>>>(x, W_in, b_in, b0, b1, b2,
                                          w0t, w1t, w2t, wtrw, out);
}
